// ContextVectorNN_86990267613475
// MI455X (gfx1250) — matrix operand order repaired and checked by reading, not compiled, not run on hardware
//
#include <hip/hip_runtime.h>
#include <hip/hip_bf16.h>
#include <math.h>

// Sizes from the reference
#define Bz   16
#define Sz   4096
#define Fz   512      // F_TEXT
#define Hz   512      // NUM_HIDDEN
#define NIN  1025     // NUM_INPUTS (512 text + 512 summary + 1 coverage)
#define STILE 128     // s positions per workgroup in the GEMM kernel
#define KCH   32      // K chunk (one bf16 WMMA K-step)
#define NKCH  (Fz / KCH)
#define LDP   36      // padded LDS row stride (bf16 elems): 72B, odd bank stride

typedef __attribute__((ext_vector_type(16))) __bf16 v16bf;
typedef __attribute__((ext_vector_type(4)))  __bf16 v4bf;
typedef __attribute__((ext_vector_type(8)))  float  v8f;
typedef unsigned int u32x4 __attribute__((ext_vector_type(4)));
typedef int          i32x8 __attribute__((ext_vector_type(8)));
typedef int          i32x4 __attribute__((ext_vector_type(4)));

#if __has_builtin(__builtin_amdgcn_tensor_load_to_lds) && \
    __has_builtin(__builtin_amdgcn_s_wait_tensorcnt)
#define USE_TDM 1
#else
#define USE_TDM 0
#endif

#if USE_TDM
// Issue one TDM 2D-tile load: ts tile [STILE rows x KCH cols] fp32 -> LDS.
// D# per CDNA5 ISA §8.3/8.4. Per-wave instruction, EXEC ignored.
__device__ __forceinline__ void tdm_load_tile(const float* gptr,
                                              unsigned int lds_off) {
  const unsigned long long ga = (unsigned long long)(const void*)gptr;
  u32x4 g0 = {};
  g0[0] = 1u;                                              // count=1 (valid)
  g0[1] = lds_off;                                         // lds_addr
  g0[2] = (unsigned int)ga;                                // global_addr lo
  g0[3] = ((unsigned int)(ga >> 32) & 0x01FFFFFFu) | 0x80000000u;  // hi | type=2
  i32x8 g1 = {};
  g1[0] = (int)(2u << 16);                    // data_size = 4 bytes
  g1[1] = (int)((Fz & 0xFFFFu) << 16);        // tensor_dim0 = 512 (lo16)
  g1[2] = (int)((Sz & 0xFFFFu) << 16);        // dim0 hi16=0 | tensor_dim1 lo16
  g1[3] = (int)((unsigned)KCH << 16);         // dim1 hi16=0 | tile_dim0 = 32
  g1[4] = (int)STILE;                         // tile_dim1 = 128, tile_dim2 = 0
  g1[5] = (int)Fz;                            // tensor_dim0_stride = 512
  g1[6] = 0;
  g1[7] = 0;
  i32x4 gz = {};
#if __clang_major__ >= 23
  i32x8 gz8 = {};
  __builtin_amdgcn_tensor_load_to_lds(g0, g1, gz, gz, gz8, 0);
#else
  __builtin_amdgcn_tensor_load_to_lds(g0, g1, gz, gz, 0);
#endif
}
#endif

// ---------------------------------------------------------------------------
// Kernel A: pre-swizzle W1_text into bf16 A-fragments (done once, 512 KB,
// stays L2-resident). Layout: frag[((w*NKCH + kc)*32 + lane)*16 + e], where
// 16-bit A layout is lanes 0-15: K={0..7,16..23}, lanes 16-31: K={8..15,24..31}.
// ---------------------------------------------------------------------------
__global__ void w1swizzle_kernel(const float* __restrict__ W1,
                                 __bf16* __restrict__ frag) {
  const int idx  = blockIdx.x * blockDim.x + threadIdx.x;  // 0..16383
  const int lane = idx & 31;
  const int kc   = (idx >> 5) & (NKCH - 1);
  const int w    = idx >> 9;                               // 0..31 (h tile)
  const int m    = lane & 15;
  const int kh   = lane >> 4;
  const float* src = W1 + (size_t)(w * 16 + m) * NIN + kc * KCH + kh * 8;
  __bf16* dst = frag + (size_t)idx * 16;
#pragma unroll
  for (int e = 0; e < 8; ++e) {
    dst[e]     = (__bf16)src[e];        // K = base + 0..7
    dst[8 + e] = (__bf16)src[16 + e];   // K = base + 16..23
  }
}

// ---------------------------------------------------------------------------
// Kernel B: per-batch summary bias  sbias[b][h] = b1[h] + W1[h,512:1024]·summ[b]
// ---------------------------------------------------------------------------
__global__ void sbias_kernel(const float* __restrict__ W1,
                             const float* __restrict__ b1,
                             const float* __restrict__ summ,
                             float* __restrict__ sbias) {
  const int idx = blockIdx.x * blockDim.x + threadIdx.x;   // 0..8191
  const int b = idx >> 9;
  const int h = idx & 511;
  const float* wr = W1 + (size_t)h * NIN + 512;
  const float* sm = summ + b * 512;
  float acc = b1[h];
  for (int j = 0; j < 512; ++j) acc = fmaf(wr[j], sm[j], acc);
  sbias[idx] = acc;   // idx == b*512 + h
}

// ---------------------------------------------------------------------------
// Kernel C: fused  scores[b][s] = W2 · tanh(W1_text·ts + sbias + W1_cov*cov) + b2
// 32 waves per block; wave w owns h-stripe [16w,16w+16), 8 s-subtiles of 16.
// ts tiles are DMA'd by the Tensor Data Mover into a double-buffered LDS
// staging area (DMA of chunk kc+1 overlaps convert+WMMA of chunk kc), then
// block-converted to bf16. A fragments come straight from pre-swizzled global.
// ---------------------------------------------------------------------------
__global__ __launch_bounds__(1024) void score_kernel(
    const float* __restrict__ ts, const float* __restrict__ cov,
    const float* __restrict__ W1, const float* __restrict__ W2,
    const float* __restrict__ b2p, const float* __restrict__ sbias,
    const __bf16* __restrict__ w1frag, float* __restrict__ scores) {
#if USE_TDM
  __shared__ float  ltsf[2][STILE][KCH];  // TDM dest, fp32, double buffer (32 KB)
#endif
  __shared__ __bf16 lts[STILE][LDP];      // bf16 tile, padded rows (9.2 KB)
  __shared__ float  lsc[STILE];           // cross-wave score accumulation

  const int tid  = threadIdx.x;
  const int b    = blockIdx.y;
  const int s0   = blockIdx.x * STILE;
  const int lane = tid & 31;
  const int w    = tid >> 5;       // 0..31
  const int m    = lane & 15;
  const int kh   = lane >> 4;

  if (tid < STILE) lsc[tid] = b2p[0];

  v8f acc[8] = {};                 // 8 s-subtiles of 16x16 f32

  const size_t ts_base = ((size_t)b * Sz + s0) * Fz;

#if USE_TDM
  // Prologue: kick off DMA of chunk 0.
  if (tid < 32)
    tdm_load_tile(ts + ts_base, (unsigned int)(size_t)(void*)&ltsf[0][0][0]);
#endif

  for (int kc = 0; kc < NKCH; ++kc) {
    const int k0 = kc * KCH;

#if USE_TDM
    // ---- pipelined TDM: issue chunk kc+1, then ensure chunk kc has landed ----
    if (tid < 32) {
      if (kc + 1 < NKCH) {
        tdm_load_tile(ts + ts_base + (kc + 1) * KCH,
                      (unsigned int)(size_t)(void*)&ltsf[(kc + 1) & 1][0][0]);
        __builtin_amdgcn_s_wait_tensorcnt(1);  // in-order: chunk kc complete
      } else {
        __builtin_amdgcn_s_wait_tensorcnt(0);
      }
    }
    __syncthreads();   // chunk kc visible to all; also fences last chunk's lts reads
    // ---- block-wide fp32 -> bf16 convert into padded LDS tile ----
    {
      const int si = tid >> 3;            // 0..127
      const int kq = (tid & 7) * 4;       // 0..28
      const float4 v = *(const float4*)&ltsf[kc & 1][si][kq];
      lts[si][kq]     = (__bf16)v.x;
      lts[si][kq + 1] = (__bf16)v.y;
      lts[si][kq + 2] = (__bf16)v.z;
      lts[si][kq + 3] = (__bf16)v.w;
    }
#else
    // ---- fallback: direct global->reg->LDS staging (4 elems / thread) ----
    {
      const int si = tid >> 3;
      const int kq = (tid & 7) * 4;
      const float4 v = *(const float4*)(ts + ts_base + (size_t)si * Fz + k0 + kq);
      lts[si][kq]     = (__bf16)v.x;
      lts[si][kq + 1] = (__bf16)v.y;
      lts[si][kq + 2] = (__bf16)v.z;
      lts[si][kq + 3] = (__bf16)v.w;
      if (kc + 1 < NKCH)
        __builtin_prefetch(ts + ts_base + (size_t)si * Fz + k0 + KCH + kq, 0, 1);
    }
    __syncthreads();   // (fallback needs WAR fence before restaging lts)
#endif
    __syncthreads();

    // ---- A fragment: one coalesced 32B load from pre-swizzled W1 ----
    const v16bf a =
        *(const v16bf*)(w1frag + (((size_t)w * NKCH + kc) * 32 + lane) * 16);

    // ---- 8 s-subtiles: B fragment (32x16): lane n holds col n,
    //      lanes 0-15 K={0..7,16..23}, lanes 16-31 K={8..15,24..31} ----
#pragma unroll
    for (int st = 0; st < 8; ++st) {
      const __bf16* br = &lts[st * 16 + m][kh * 8];
      const v4bf q0 = *(const v4bf*)(br + 0);
      const v4bf q1 = *(const v4bf*)(br + 4);
      const v4bf q2 = *(const v4bf*)(br + 16);
      const v4bf q3 = *(const v4bf*)(br + 20);
      v16bf bb;
#pragma unroll
      for (int i = 0; i < 4; ++i) {
        bb[i] = q0[i]; bb[4 + i] = q1[i]; bb[8 + i] = q2[i]; bb[12 + i] = q3[i];
      }
      acc[st] = __builtin_amdgcn_wmma_f32_16x16x32_bf16(
          false, a, false, bb, (short)0, acc[st], false, false);
    }
#if !USE_TDM
    // handled by the loop-top barrier in the TDM path
#endif
  }
  __syncthreads();

  // ---- epilogue: tanh + W2 reduction, fused (h never hits memory) ----
  // C layout: VGPR r -> M = r + 8*(lane>=16), N = lane&15
  float sb[8], wc[8], w2v[8];
#pragma unroll
  for (int r = 0; r < 8; ++r) {
    const int h = (w * 16) + kh * 8 + r;
    sb[r]  = sbias[b * Hz + h];
    wc[r]  = W1[(size_t)h * NIN + 1024];   // coverage column of W1
    w2v[r] = W2[h];
  }
#pragma unroll
  for (int st = 0; st < 8; ++st) {
    const int s = s0 + st * 16 + m;
    const float cv = cov[(size_t)b * Sz + s];
    float p = 0.f;
#pragma unroll
    for (int r = 0; r < 8; ++r) {
      const float hv = tanhf(acc[st][r] + sb[r] + wc[r] * cv);
      p = fmaf(w2v[r], hv, p);
    }
    p += __shfl_xor(p, 16, 32);            // combine M=0..7 with M=8..15 halves
    if (lane < 16) atomicAdd(&lsc[st * 16 + m], p);  // ds_add_f32 across waves
  }
  __syncthreads();
  if (tid < STILE) scores[(size_t)b * Sz + s0 + tid] = lsc[tid];
}

// ---------------------------------------------------------------------------
// Kernel D: per-batch softmax over S, write attention, then context GEMV
// ---------------------------------------------------------------------------
__global__ __launch_bounds__(1024) void softmax_ctx_kernel(
    const float* __restrict__ ts, const float* __restrict__ scores,
    float* __restrict__ out) {
  __shared__ float att[Sz];    // 16 KB
  __shared__ float red[1024];  // 4 KB
  const int tid = threadIdx.x;
  const int b   = blockIdx.x;
  const float* sc = scores + (size_t)b * Sz;

  float s4[4];
  float mx = -INFINITY;
#pragma unroll
  for (int k = 0; k < 4; ++k) { s4[k] = sc[tid + k * 1024]; mx = fmaxf(mx, s4[k]); }
  red[tid] = mx; __syncthreads();
  for (int off = 512; off > 0; off >>= 1) {
    if (tid < off) red[tid] = fmaxf(red[tid], red[tid + off]);
    __syncthreads();
  }
  mx = red[0]; __syncthreads();

  float lsum = 0.f;
#pragma unroll
  for (int k = 0; k < 4; ++k) {
    const float e = __expf(s4[k] - mx);
    att[tid + k * 1024] = e;
    lsum += e;
  }
  red[tid] = lsum; __syncthreads();
  for (int off = 512; off > 0; off >>= 1) {
    if (tid < off) red[tid] += red[tid + off];
    __syncthreads();
  }
  const float inv = 1.0f / red[0]; __syncthreads();

#pragma unroll
  for (int k = 0; k < 4; ++k) {
    const float a = att[tid + k * 1024] * inv;
    att[tid + k * 1024] = a;
    out[(size_t)Bz * Fz + (size_t)b * Sz + tid + k * 1024] = a;  // attention out
  }
  __syncthreads();

  // context[b][f] = sum_s att[s] * ts[b,s,f]  (f coalesced across lanes)
  const int f    = tid & 511;
  const int half = tid >> 9;
  const float* tsb = ts + (size_t)b * Sz * Fz + f;
  float c = 0.f;
  const int sbeg = half * (Sz / 2);
  for (int s = sbeg; s < sbeg + Sz / 2; ++s)
    c = fmaf(att[s], tsb[(size_t)s * Fz], c);
  red[tid] = c; __syncthreads();
  if (tid < 512) out[b * Fz + tid] = red[tid] + red[tid + 512];
}

// ---------------------------------------------------------------------------
extern "C" void kernel_launch(void* const* d_in, const int* in_sizes, int n_in,
                              void* d_out, int out_size, void* d_ws, size_t ws_size,
                              hipStream_t stream) {
  (void)in_sizes; (void)n_in; (void)out_size; (void)ws_size;
  const float* ts   = (const float*)d_in[0];
  const float* summ = (const float*)d_in[1];
  const float* cov  = (const float*)d_in[2];
  const float* W1   = (const float*)d_in[3];
  const float* b1   = (const float*)d_in[4];
  const float* W2   = (const float*)d_in[5];
  const float* b2   = (const float*)d_in[6];
  float* out    = (float*)d_out;
  float*  sbias  = (float*)d_ws;                    // [16*512]  (32 KB)
  float*  scores = sbias + Bz * Hz;                 // [16*4096] (256 KB)
  __bf16* w1frag = (__bf16*)(scores + Bz * Sz);     // [512*512] bf16 (512 KB)

  w1swizzle_kernel<<<64, 256, 0, stream>>>(W1, w1frag);
  sbias_kernel<<<32, 256, 0, stream>>>(W1, b1, summ, sbias);
  score_kernel<<<dim3(Sz / STILE, Bz), 1024, 0, stream>>>(ts, cov, W1, W2, b2,
                                                          sbias, w1frag, scores);
  softmax_ctx_kernel<<<Bz, 1024, 0, stream>>>(ts, scores, out);
}
